// Kuramoto_Solver_3959959847449
// MI455X (gfx1250) — matrix operand order repaired and checked by reading, not compiled, not run on hardware
//
#include <hip/hip_runtime.h>
#include <hip/hip_bf16.h>
#include <math.h>

#define B_    8
#define N_    2048
#define C_    256
#define G4    (C_/4)     // 64 oscillator groups
#define RANK_ 16
#define EPS_    1e-6f
#define GN_EPS_ 1e-5f

typedef __bf16 bf16;
typedef __attribute__((ext_vector_type(8)))  __bf16 v8bf;
typedef __attribute__((ext_vector_type(16))) __bf16 v16bf;
typedef __attribute__((ext_vector_type(8)))  float  v8f;

// sched_group_barrier masks
#define SGB_MFMA   0x008
#define SGB_VMEMRD 0x020
#define SGB_DSRD   0x100
#define SGB_DSWR   0x200

// ---------------------------------------------------------------------------
// 1) GroupNorm over y[B,C,N] (group = 4 channels x N) fused with transpose to
//    [B,N,C] and sphere-normalization of each 4-channel group (they coincide).
// ---------------------------------------------------------------------------
__global__ void gn_sphere_y(const float* __restrict__ y,
                            const float* __restrict__ gw,
                            const float* __restrict__ gb,
                            float* __restrict__ ys) {
  const int bid = blockIdx.x;           // b*G4 + g
  const int b = bid >> 6, g = bid & 63;
  const float* yp = y + ((size_t)b * C_ + 4 * g) * N_;   // 4 rows of N
  const int tid = threadIdx.x;

  float s = 0.f, ss = 0.f;
  for (int i = tid; i < 4 * N_; i += 256) { float v = yp[i]; s += v; ss += v * v; }
  __shared__ float sh_s[256], sh_q[256];
  sh_s[tid] = s; sh_q[tid] = ss;
  __syncthreads();
  for (int off = 128; off > 0; off >>= 1) {
    if (tid < off) { sh_s[tid] += sh_s[tid + off]; sh_q[tid] += sh_q[tid + off]; }
    __syncthreads();
  }
  const float inv = 1.f / (4.f * N_);
  const float mu = sh_s[0] * inv;
  const float var = sh_q[0] * inv - mu * mu;
  const float rstd = rsqrtf(var + GN_EPS_);
  const float w0 = gw[4*g+0], w1 = gw[4*g+1], w2 = gw[4*g+2], w3 = gw[4*g+3];
  const float b0 = gb[4*g+0], b1 = gb[4*g+1], b2 = gb[4*g+2], b3 = gb[4*g+3];

  for (int n = tid; n < N_; n += 256) {
    float v0 = (yp[0*N_ + n] - mu) * rstd * w0 + b0;
    float v1 = (yp[1*N_ + n] - mu) * rstd * w1 + b1;
    float v2 = (yp[2*N_ + n] - mu) * rstd * w2 + b2;
    float v3 = (yp[3*N_ + n] - mu) * rstd * w3 + b3;
    float s2 = fmaxf(v0*v0 + v1*v1 + v2*v2 + v3*v3, EPS_);
    float r = rsqrtf(s2);
    float4 o = make_float4(v0*r, v1*r, v2*r, v3*r);
    *reinterpret_cast<float4*>(ys + ((size_t)b * N_ + n) * C_ + 4 * g) = o;
  }
}

// ---------------------------------------------------------------------------
// 2) x0 = map_to_sphere(x)
// ---------------------------------------------------------------------------
__global__ void sphere_x0(const float* __restrict__ x, float* __restrict__ xc) {
  size_t idx = (size_t)blockIdx.x * 256 + threadIdx.x;   // B*N*G4 threads
  float4 v = reinterpret_cast<const float4*>(x)[idx];
  float s2 = fmaxf(v.x*v.x + v.y*v.y + v.z*v.z + v.w*v.w, EPS_);
  float r = rsqrtf(s2);
  v.x *= r; v.y *= r; v.z *= r; v.w *= r;
  reinterpret_cast<float4*>(xc)[idx] = v;
}

// ---------------------------------------------------------------------------
// 3) K[b,n,m] = 0.5*(sc[b,n,m]+sc[b,m,n]) * sigmoid(U[n]·U[m])  -> bf16
// ---------------------------------------------------------------------------
__global__ void build_K(const float* __restrict__ sc,
                        const float* __restrict__ U,
                        bf16* __restrict__ Kbf) {
  const int tx = threadIdx.x, ty = threadIdx.y;
  const int m0 = blockIdx.x * 16, n0 = blockIdx.y * 16, b = blockIdx.z;
  __shared__ float Un[16][16], Um[16][16], T[16][17];
  Un[ty][tx] = U[(n0 + ty) * RANK_ + tx];
  Um[ty][tx] = U[(m0 + ty) * RANK_ + tx];
  T[ty][tx]  = sc[((size_t)b * N_ + m0 + ty) * N_ + n0 + tx];
  __syncthreads();
  float s = 0.f;
#pragma unroll
  for (int r = 0; r < 16; ++r) s += Un[ty][r] * Um[tx][r];
  float sig = 1.f / (1.f + __expf(-s));
  float a = sc[((size_t)b * N_ + n0 + ty) * N_ + m0 + tx];
  float kv = 0.5f * (a + T[tx][ty]) * sig;
  Kbf[((size_t)b * N_ + n0 + ty) * N_ + m0 + tx] = (bf16)kv;
}

// ---------------------------------------------------------------------------
// 4a) xc[B,N,C] f32  ->  xcT[B,C,N] bf16  (LDS 32x32 tile transpose)
// ---------------------------------------------------------------------------
__global__ void xc_to_bf16T(const float* __restrict__ xc, bf16* __restrict__ xcT) {
  __shared__ float tile[32][33];
  const int tx = threadIdx.x, ty = threadIdx.y;
  const int n0 = blockIdx.x * 32, c0 = blockIdx.y * 32, b = blockIdx.z;
#pragma unroll
  for (int i = 0; i < 4; ++i)
    tile[ty + 8*i][tx] = xc[((size_t)(b * N_ + n0 + ty + 8*i)) * C_ + c0 + tx];
  __syncthreads();
#pragma unroll
  for (int i = 0; i < 4; ++i)
    xcT[((size_t)(b * C_ + c0 + ty + 8*i)) * N_ + n0 + tx] = (bf16)tile[tx][ty + 8*i];
}

// ---------------------------------------------------------------------------
// 4b) coupling[b] = K[b] (NxN bf16) @ xc[b] (NxC bf16) -> f32
//
// Block: 256 threads = 8 waves -> 64 rows x 256 cols.
// B chunk (32k x 256c) double-buffered in LDS; B tile register-double-buffered
// (load tile t+1 before WMMA t -> s_wait_dscnt <= 2 instead of 0); A operand
// pipelined one k-chunk ahead. sched_group_barrier enforces the cadence.
// ---------------------------------------------------------------------------
#define LDS_STRIDE 40   // halves per column (32 data + 8 pad) -> conflict-free

__device__ __forceinline__ v16bf ld_btile(const bf16* p) {
  v8bf lo = *reinterpret_cast<const v8bf*>(p);
  v8bf hi = *reinterpret_cast<const v8bf*>(p + 16);
  return __builtin_shufflevector(lo, hi,
         0,1,2,3,4,5,6,7,8,9,10,11,12,13,14,15);
}

__device__ __forceinline__ v16bf ld_a(const bf16* p) {
  v8bf lo = *reinterpret_cast<const v8bf*>(p);        // K base..base+7
  v8bf hi = *reinterpret_cast<const v8bf*>(p + 16);   // K base+16..base+23
  return __builtin_shufflevector(lo, hi,
         0,1,2,3,4,5,6,7,8,9,10,11,12,13,14,15);
}

__global__ void __launch_bounds__(256)
coupling_gemm(const bf16* __restrict__ Kbf, const bf16* __restrict__ xcT,
              float* __restrict__ coup) {
  __shared__ __align__(16) bf16 lds[2][C_ * LDS_STRIDE];   // 2 x 20 KB

  const int tid  = threadIdx.x;
  const int lane = tid & 31;
  const int wave = tid >> 5;
  const int b    = blockIdx.x >> 5;             // 32 row-blocks (64 rows) per batch
  const int R0   = (blockIdx.x & 31) * 64;
  const int rg   = wave >> 1;                   // 0..3 : 16-row group
  const int h    = wave & 1;                    // 0..1 : 128-col half
  const int l16  = lane & 15;
  const int hi   = lane >> 4;

  // A layout (16-bit A 16x32, ISA 7.12.2): lanes 0-15 row=l, K={0..7,16..23};
  // lanes 16-31 row=l-16, K={8..15,24..31}.
  const bf16* aRow = Kbf + ((size_t)(b * N_ + R0 + 16 * rg + l16)) * N_ + 8 * hi;
  // staging: thread tid owns column tid of xcT (k contiguous)
  const bf16* bCol = xcT + ((size_t)(b * C_ + tid)) * N_;
  // per-wave LDS read base (B layout: lane&15 = col, lanes 0-15 K={0..7,16..23}, 16-31 K={8..15,24..31})
  const int lbOff = (128 * h + l16) * LDS_STRIDE + 8 * hi;

  v8f acc[8];
#pragma unroll
  for (int t = 0; t < 8; ++t) acc[t] = v8f{};

  // ---- prologue: stage chunk 0 into lds[0]; load A chunk 0 ----
  {
    const float4* p = reinterpret_cast<const float4*>(bCol);
    float4 s0 = p[0], s1 = p[1], s2 = p[2], s3 = p[3];    // 64 B = 32 halves
    float4* d = reinterpret_cast<float4*>(&lds[0][tid * LDS_STRIDE]);
    d[0] = s0; d[1] = s1; d[2] = s2; d[3] = s3;
  }
  v16bf Acur = ld_a(aRow);
  __syncthreads();

  // ---- main loop: compute chunk kk from lds[buf], stage kk+32 into lds[buf^1]
  for (int kk = 0; kk < N_ - 32; kk += 32) {
    const int buf = (kk >> 5) & 1;
    __builtin_amdgcn_sched_barrier(0);

    // next B chunk global prefetch (4 b128), then next A chunk (2 b128);
    // kk+32 <= N_-32 always, so A prefetch needs no guard.
    const float4* p = reinterpret_cast<const float4*>(bCol + kk + 32);
    float4 s0 = p[0], s1 = p[1], s2 = p[2], s3 = p[3];
    v8bf nlo = *reinterpret_cast<const v8bf*>(aRow + kk + 32);
    v8bf nhi = *reinterpret_cast<const v8bf*>(aRow + kk + 48);

    // 8 WMMAs; B tile register-double-buffered (load t+1 before wmma t)
    const bf16* lb = &lds[buf][lbOff];
    v16bf Bcur = ld_btile(lb);
#pragma unroll
    for (int t = 0; t < 8; ++t) {
      if (t < 7) {
        v16bf Bnext = ld_btile(lb + 16 * (t + 1) * LDS_STRIDE);
        acc[t] = __builtin_amdgcn_wmma_f32_16x16x32_bf16(
            false, Acur, false, Bcur, (short)0, acc[t], false, false);
        Bcur = Bnext;
      } else {
        acc[t] = __builtin_amdgcn_wmma_f32_16x16x32_bf16(
            false, Acur, false, Bcur, (short)0, acc[t], false, false);
      }
    }

    // commit staged chunk to the other buffer (4 ds_store_b128)
    {
      float4* d = reinterpret_cast<float4*>(&lds[buf ^ 1][tid * LDS_STRIDE]);
      d[0] = s0; d[1] = s1; d[2] = s2; d[3] = s3;
    }
    // rotate A pipeline
    Acur = __builtin_shufflevector(nlo, nhi,
           0,1,2,3,4,5,6,7,8,9,10,11,12,13,14,15);

    // --- cadence: tile0 loads; then per-tile [2 ds_read, wmma] with the 4
    //     B-chunk prefetches (t=0..3), 2 A-next loads (t=4) and 4 ds_writes
    //     (t=5,6) distributed between WMMAs.
    __builtin_amdgcn_sched_group_barrier(SGB_DSRD, 2, 0);      // tile0
#pragma unroll
    for (int t = 0; t < 4; ++t) {
      __builtin_amdgcn_sched_group_barrier(SGB_DSRD,   2, 0);  // tile t+1
      __builtin_amdgcn_sched_group_barrier(SGB_MFMA,   1, 0);  // wmma t
      __builtin_amdgcn_sched_group_barrier(SGB_VMEMRD, 1, 0);  // B prefetch
    }
    __builtin_amdgcn_sched_group_barrier(SGB_DSRD,   2, 0);    // tile5
    __builtin_amdgcn_sched_group_barrier(SGB_MFMA,   1, 0);    // wmma4
    __builtin_amdgcn_sched_group_barrier(SGB_VMEMRD, 2, 0);    // A-next
    __builtin_amdgcn_sched_group_barrier(SGB_DSRD,   2, 0);    // tile6
    __builtin_amdgcn_sched_group_barrier(SGB_MFMA,   1, 0);    // wmma5
    __builtin_amdgcn_sched_group_barrier(SGB_DSWR,   2, 0);    // stage store
    __builtin_amdgcn_sched_group_barrier(SGB_DSRD,   2, 0);    // tile7
    __builtin_amdgcn_sched_group_barrier(SGB_MFMA,   1, 0);    // wmma6
    __builtin_amdgcn_sched_group_barrier(SGB_DSWR,   2, 0);    // stage store
    __builtin_amdgcn_sched_group_barrier(SGB_MFMA,   1, 0);    // wmma7
    __builtin_amdgcn_sched_barrier(0);
    __syncthreads();
  }

  // ---- epilogue: last chunk (staged into lds[1]; Acur already prefetched) ----
  {
    const bf16* lb = &lds[1][lbOff];
    v16bf Bcur = ld_btile(lb);
#pragma unroll
    for (int t = 0; t < 8; ++t) {
      if (t < 7) {
        v16bf Bnext = ld_btile(lb + 16 * (t + 1) * LDS_STRIDE);
        acc[t] = __builtin_amdgcn_wmma_f32_16x16x32_bf16(
            false, Acur, false, Bcur, (short)0, acc[t], false, false);
        Bcur = Bnext;
      } else {
        acc[t] = __builtin_amdgcn_wmma_f32_16x16x32_bf16(
            false, Acur, false, Bcur, (short)0, acc[t], false, false);
      }
    }
  }

  // C/D layout: VGPR i -> row 8*hi + i, column = lane&15 (coalesced 64B rows)
#pragma unroll
  for (int t = 0; t < 8; ++t) {
    const int col = 128 * h + 16 * t + l16;
    float* op = coup + ((size_t)(b * N_ + R0 + 16 * rg + 8 * hi)) * C_ + col;
#pragma unroll
    for (int i = 0; i < 8; ++i)
      op[(size_t)i * C_] = acc[t][i];
  }
}

// ---------------------------------------------------------------------------
// 4c) fused Kuramoto step
// ---------------------------------------------------------------------------
__global__ void step_pointwise(float* __restrict__ xc,
                               const float* __restrict__ coup,
                               const float* __restrict__ ys,
                               const float* __restrict__ omega,
                               const float* __restrict__ gamma_p,
                               float* __restrict__ out_q) {
  size_t idx = (size_t)blockIdx.x * 256 + threadIdx.x;   // B*N*G4 threads
  const int g = (int)(idx & 63);
  float4 x  = reinterpret_cast<const float4*>(xc)[idx];
  float4 cp = reinterpret_cast<const float4*>(coup)[idx];
  float4 yv = reinterpret_cast<const float4*>(ys)[idx];
  float fx = cp.x + yv.x, fy = cp.y + yv.y, fz = cp.z + yv.z, fw = cp.w + yv.w;
  float sim = x.x*fx + x.y*fy + x.z*fz + x.w*fw;
  float px = fx - sim*x.x, py = fy - sim*x.y, pz = fz - sim*x.z, pw = fw - sim*x.w;
  float w0 = fabsf(omega[2*g]), w1 = fabsf(omega[2*g + 1]);
  float d0 =  w0 * x.y + px;
  float d1 = -w0 * x.x + py;
  float d2 =  w1 * x.w + pz;
  float d3 = -w1 * x.z + pw;
  float gm = *gamma_p;
  float n0 = x.x + gm*d0, n1 = x.y + gm*d1, n2 = x.z + gm*d2, n3 = x.w + gm*d3;
  float s2 = fmaxf(n0*n0 + n1*n1 + n2*n2 + n3*n3, EPS_);
  float r = rsqrtf(s2);
  float4 o = make_float4(n0*r, n1*r, n2*r, n3*r);
  reinterpret_cast<float4*>(xc)[idx]    = o;
  reinterpret_cast<float4*>(out_q)[idx] = o;
}

// ---------------------------------------------------------------------------
// Workspace: Kbf 64MiB | ys 16MiB | xc 16MiB | coup 16MiB | xcT 8MiB  (~126MiB)
// ---------------------------------------------------------------------------
extern "C" void kernel_launch(void* const* d_in, const int* in_sizes, int n_in,
                              void* d_out, int out_size, void* d_ws, size_t ws_size,
                              hipStream_t stream) {
  (void)in_sizes; (void)n_in; (void)ws_size;
  const float* x     = (const float*)d_in[0];
  const float* y     = (const float*)d_in[1];
  const float* sc    = (const float*)d_in[2];
  const float* U     = (const float*)d_in[3];
  const float* omega = (const float*)d_in[4];
  const float* gw    = (const float*)d_in[5];
  const float* gb    = (const float*)d_in[6];
  const float* gamma = (const float*)d_in[7];
  float* out = (float*)d_out;

  char* w = (char*)d_ws;
  const size_t SZ_K  = (size_t)B_ * N_ * N_ * sizeof(bf16);   // 64 MiB
  const size_t SZ_F  = (size_t)B_ * N_ * C_ * sizeof(float);  // 16 MiB
  bf16*  Kbf  = (bf16*)(w);
  float* ys   = (float*)(w + SZ_K);
  float* xc   = (float*)(w + SZ_K + SZ_F);
  float* coup = (float*)(w + SZ_K + 2 * SZ_F);
  bf16*  xcT  = (bf16*)(w + SZ_K + 3 * SZ_F);

  const int Q = out_size / (B_ * N_ * C_);   // = 8 for this problem

  gn_sphere_y<<<B_ * G4, 256, 0, stream>>>(y, gw, gb, ys);
  sphere_x0<<<(B_ * N_ * G4) / 256, 256, 0, stream>>>(x, xc);
  build_K<<<dim3(N_ / 16, N_ / 16, B_), dim3(16, 16), 0, stream>>>(sc, U, Kbf);

  for (int q = 0; q < Q; ++q) {
    xc_to_bf16T<<<dim3(N_ / 32, C_ / 32, B_), dim3(32, 8), 0, stream>>>(xc, xcT);
    coupling_gemm<<<B_ * (N_ / 64), 256, 0, stream>>>(Kbf, xcT, coup);
    step_pointwise<<<(B_ * N_ * G4) / 256, 256, 0, stream>>>(
        xc, coup, ys, omega, gamma, out + (size_t)q * B_ * N_ * C_);
  }
}
